// MPNNet_Parametric_65438121722209
// MI455X (gfx1250) — matrix operand order repaired and checked by reading, compiled, not run on hardware
//
#include <hip/hip_runtime.h>
#include <hip/hip_bf16.h>
#include <math.h>

#define N_NODES 15000
#define N_EDGES 30000
#define NUM_GRAPHS 600
#define N_STEMS 6000
#define N_JBONDS 3000
#define DIM 64
#define NUM_FEAT 14
#define NOUT_STEM 105

typedef __bf16 bf16_t;
typedef __attribute__((ext_vector_type(16))) __bf16 v16bf;
typedef __attribute__((ext_vector_type(8)))  __bf16 bf16x8;
typedef __attribute__((ext_vector_type(8)))  float   v8f;

union V16 { v16bf v; bf16x8 h[2]; };

__device__ __forceinline__ float lrelu(float x) { return x > 0.f ? x : 0.01f * x; }
__device__ __forceinline__ float sigm(float x)  { return 1.f / (1.f + expf(-x)); }

// ---- float <-> order-preserving uint (for atomic segment-max) ----
__device__ __forceinline__ unsigned f2ord(float f) {
    unsigned u = __float_as_uint(f);
    return (u & 0x80000000u) ? ~u : (u | 0x80000000u);
}
__device__ __forceinline__ float ord2f(unsigned v) {
    return (v & 0x80000000u) ? __uint_as_float(v ^ 0x80000000u) : __uint_as_float(~v);
}

// ---- D = A[16x64](bf16) @ B (16 cols of row-major [C,64] weight, bf16) via 2 chained WMMA ----
// A layout (per lane): row m=lane&15, h[0]=K{hi8..hi8+7}, h[1]=K{16+hi8..23+hi8}, chunk2 +32.
// B layout (per lane): col n=lane&15, h[0]=K{hi8..hi8+7}, h[1]=K{16+hi8..23+hi8}, chunk2 +32.
__device__ __forceinline__ v8f wmma_tile(const V16& a0, const V16& a1,
                                         const bf16_t* __restrict__ W,
                                         int cbase, int n, int bhalf) {
    V16 b0, b1;
    const bf16_t* brow = W + (size_t)(cbase + n) * DIM;
    b0.h[0] = *(const bf16x8*)(brow + bhalf / 2);
    b0.h[1] = *(const bf16x8*)(brow + 16 + bhalf / 2);
    b1.h[0] = *(const bf16x8*)(brow + 32 + bhalf / 2);
    b1.h[1] = *(const bf16x8*)(brow + 48 + bhalf / 2);
    v8f d = {0.f, 0.f, 0.f, 0.f, 0.f, 0.f, 0.f, 0.f};
    d = __builtin_amdgcn_wmma_f32_16x16x32_bf16(false, a0.v, false, b0.v, (short)0, d, false, false);
    d = __builtin_amdgcn_wmma_f32_16x16x32_bf16(false, a1.v, false, b1.v, (short)0, d, false, false);
    return d;
}

// ================= utility kernels =================
__global__ void zero_f32(float* p, int n) {
    int t = blockIdx.x * blockDim.x + threadIdx.x;
    if (t < n) p[t] = 0.f;
}

__global__ void to_bf16(const float* __restrict__ s, bf16_t* __restrict__ d, int n) {
    int t = blockIdx.x * blockDim.x + threadIdx.x;
    if (t < n) d[t] = (bf16_t)s[t];
}

// ================= node/edge encoders =================
__global__ void lin0_kernel(const float* __restrict__ x, const float* __restrict__ w,
                            const float* __restrict__ b, float* __restrict__ nodeF,
                            bf16_t* __restrict__ nodeBf) {
    int t = blockIdx.x * blockDim.x + threadIdx.x;
    if (t >= N_NODES * DIM) return;
    int node = t / DIM, o = t % DIM;
    const float* xr = x + (size_t)node * NUM_FEAT;
    const float* wr = w + (size_t)o * NUM_FEAT;
    float acc = b[o];
#pragma unroll
    for (int f = 0; f < NUM_FEAT; ++f) acc += xr[f] * wr[f];
    acc = lrelu(acc);
    nodeF[t]  = acc;
    nodeBf[t] = (bf16_t)acc;
}

__global__ void he_kernel(const float* __restrict__ ea, const float* __restrict__ w,
                          const float* __restrict__ b, bf16_t* __restrict__ heBf) {
    int t = blockIdx.x * blockDim.x + threadIdx.x;
    if (t >= N_EDGES * DIM) return;
    int e = t / DIM, o = t % DIM;
    const float* er = ea + (size_t)e * 4;
    const float* wr = w + (size_t)o * 4;
    float acc = b[o] + er[0]*wr[0] + er[1]*wr[1] + er[2]*wr[2] + er[3]*wr[3];
    heBf[t] = (bf16_t)lrelu(acc);
}

__global__ void deg_kernel(const int* __restrict__ eidx, float* __restrict__ deg) {
    int e = blockIdx.x * blockDim.x + threadIdx.x;
    if (e >= N_EDGES) return;
    atomicAdd(&deg[eidx[N_EDGES + e]], 1.0f);
}

// ================= fused We-recompute + edge matvec + scatter =================
// Per wave: 16-edge tile.  acc[e,o] = sum_i out[src[e],i] * (WMMA(he,en2w)[e, i*64+o] + en2_b[i*64+o])
__global__ __launch_bounds__(128)
void msg_kernel(const bf16_t* __restrict__ he, const bf16_t* __restrict__ en2w,
                const float* __restrict__ en2b, const float* __restrict__ nodeF,
                const int* __restrict__ eidx, float* __restrict__ agg) {
    __shared__ float sS[4][16][DIM];     // gathered src features, one 16x64 tile per wave
    __shared__ float sB[DIM * DIM];      // en2_b staged (16 KB)
    const int tid  = threadIdx.x;
    const int wv   = tid >> 5;
    const int lane = tid & 31;
    const int tile = blockIdx.x * 4 + wv;
    const bool active = (tile < N_EDGES / 16);
    const int e0 = active ? tile * 16 : 0;

    for (int i = tid; i < DIM * DIM; i += 128) sB[i] = en2b[i];
    {
        int m = lane & 15, half = lane >> 4;
        int src = eidx[e0 + m];
        const float4* row = reinterpret_cast<const float4*>(nodeF + (size_t)src * DIM + half * 32);
        float* dst = &sS[wv][m][half * 32];
#pragma unroll
        for (int q = 0; q < 8; ++q) {
            float4 v = row[q];
            dst[q * 4 + 0] = v.x; dst[q * 4 + 1] = v.y;
            dst[q * 4 + 2] = v.z; dst[q * 4 + 3] = v.w;
        }
    }
    __syncthreads();
    if (!active) return;

    const int m     = lane & 15;
    const int hi8   = (lane >> 4) * 8;
    const int mbase = (lane >> 4) * 8;
    const int n     = lane & 15;
    const int bhalf = (lane >> 4) * 16;

    V16 a0, a1;
    {
        const bf16_t* hr = he + (size_t)(e0 + m) * DIM;
        a0.h[0] = *(const bf16x8*)(hr + hi8);
        a0.h[1] = *(const bf16x8*)(hr + 16 + hi8);
        a1.h[0] = *(const bf16x8*)(hr + 32 + hi8);
        a1.h[1] = *(const bf16x8*)(hr + 48 + hi8);
    }
    v8f acc[4];
#pragma unroll
    for (int ob = 0; ob < 4; ++ob)
        acc[ob] = (v8f){0.f, 0.f, 0.f, 0.f, 0.f, 0.f, 0.f, 0.f};

    for (int i = 0; i < DIM; ++i) {
        float s[8];
#pragma unroll
        for (int r = 0; r < 8; ++r) s[r] = sS[wv][r + mbase][i];
#pragma unroll
        for (int ob = 0; ob < 4; ++ob) {
            int cbase = i * DIM + ob * 16;
            v8f d = wmma_tile(a0, a1, en2w, cbase, n, bhalf);
            float bc = sB[cbase + n];
#pragma unroll
            for (int r = 0; r < 8; ++r) acc[ob][r] += s[r] * (d[r] + bc);
        }
    }
    int dstm[8];
#pragma unroll
    for (int r = 0; r < 8; ++r) dstm[r] = eidx[N_EDGES + e0 + r + mbase];
#pragma unroll
    for (int ob = 0; ob < 4; ++ob)
#pragma unroll
        for (int r = 0; r < 8; ++r)
            atomicAdd(&agg[(size_t)dstm[r] * DIM + ob * 16 + n], acc[ob][r]);
}

// ================= m = lrelu(agg/deg + out@root_w + conv_b) =================
__global__ void m_kernel(const float* __restrict__ agg, const float* __restrict__ deg,
                         const float* __restrict__ nodeF, const float* __restrict__ rootw,
                         const float* __restrict__ convb, bf16_t* __restrict__ mBf) {
    int t = blockIdx.x * blockDim.x + threadIdx.x;
    if (t >= N_NODES * DIM) return;
    int node = t / DIM, o = t % DIM;
    float acc = agg[t] / fmaxf(deg[node], 1.f) + convb[o];
    const float* hr = nodeF + (size_t)node * DIM;
#pragma unroll 8
    for (int i = 0; i < DIM; ++i) acc += hr[i] * rootw[i * DIM + o];
    mBf[t] = (bf16_t)lrelu(acc);
}

// ================= GRU cell via WMMA (gi = m@W_ih^T, gh = h@W_hh^T) =================
__global__ __launch_bounds__(128)
void gru_kernel(const bf16_t* __restrict__ mBf, const bf16_t* __restrict__ hBf,
                const bf16_t* __restrict__ wih, const bf16_t* __restrict__ whh,
                const float* __restrict__ bih, const float* __restrict__ bhh,
                float* __restrict__ nodeF, bf16_t* __restrict__ nodeBf) {
    const int tid  = threadIdx.x;
    const int wv   = tid >> 5;
    const int lane = tid & 31;
    const int tile = blockIdx.x * 4 + wv;
    const int NT = (N_NODES + 15) / 16;
    if (tile >= NT) return;

    const int m     = lane & 15;
    const int hi8   = (lane >> 4) * 8;
    const int mbase = (lane >> 4) * 8;
    const int n     = lane & 15;
    const int bhalf = (lane >> 4) * 16;

    int rowA = tile * 16 + m;
    if (rowA > N_NODES - 1) rowA = N_NODES - 1;
    V16 am0, am1, ah0, ah1;
    {
        const bf16_t* mr = mBf + (size_t)rowA * DIM;
        const bf16_t* hr = hBf + (size_t)rowA * DIM;
        am0.h[0] = *(const bf16x8*)(mr + hi8);       am0.h[1] = *(const bf16x8*)(mr + 16 + hi8);
        am1.h[0] = *(const bf16x8*)(mr + 32 + hi8);  am1.h[1] = *(const bf16x8*)(mr + 48 + hi8);
        ah0.h[0] = *(const bf16x8*)(hr + hi8);       ah0.h[1] = *(const bf16x8*)(hr + 16 + hi8);
        ah1.h[0] = *(const bf16x8*)(hr + 32 + hi8);  ah1.h[1] = *(const bf16x8*)(hr + 48 + hi8);
    }
    int rowC[8];
#pragma unroll
    for (int r = 0; r < 8; ++r) rowC[r] = tile * 16 + r + mbase;

#pragma unroll
    for (int ob = 0; ob < 4; ++ob) {
        int col = ob * 16 + n;
        v8f ri = wmma_tile(am0, am1, wih,       ob * 16, n, bhalf);
        v8f rh = wmma_tile(ah0, ah1, whh,       ob * 16, n, bhalf);
        v8f zi = wmma_tile(am0, am1, wih,  64 + ob * 16, n, bhalf);
        v8f zh = wmma_tile(ah0, ah1, whh,  64 + ob * 16, n, bhalf);
        v8f ni = wmma_tile(am0, am1, wih, 128 + ob * 16, n, bhalf);
        v8f nh = wmma_tile(ah0, ah1, whh, 128 + ob * 16, n, bhalf);
        float b_ri = bih[col],        b_rh = bhh[col];
        float b_zi = bih[64 + col],   b_zh = bhh[64 + col];
        float b_ni = bih[128 + col],  b_nh = bhh[128 + col];
#pragma unroll
        for (int r = 0; r < 8; ++r) {
            float rg = sigm(ri[r] + b_ri + rh[r] + b_rh);
            float zg = sigm(zi[r] + b_zi + zh[r] + b_zh);
            float ng = tanhf(ni[r] + b_ni + rg * (nh[r] + b_nh));
            int node = rowC[r];
            if (node < N_NODES) {
                size_t idx = (size_t)node * DIM + col;
                float hold = nodeF[idx];
                float hnew = (1.f - zg) * ng + zg * hold;
                nodeF[idx]  = hnew;
                nodeBf[idx] = (bf16_t)hnew;
            }
        }
    }
}

// ================= prediction heads =================
__global__ void head_hid_kernel(const float* __restrict__ nodeF, const int* __restrict__ idx,
                                const float* __restrict__ w, const float* __restrict__ b,
                                float* __restrict__ hid, int rows) {
    int t = blockIdx.x * blockDim.x + threadIdx.x;
    if (t >= rows * DIM) return;
    int rr = t / DIM, o = t % DIM;
    const float* xr = nodeF + (size_t)idx[rr] * DIM;
    const float* wr = w + (size_t)o * DIM;
    float acc = b[o];
#pragma unroll 8
    for (int i = 0; i < DIM; ++i) acc += xr[i] * wr[i];
    hid[t] = lrelu(acc);
}

__global__ void stem_out_kernel(const float* __restrict__ hid, const float* __restrict__ w,
                                const float* __restrict__ b, float* __restrict__ out) {
    int t = blockIdx.x * blockDim.x + threadIdx.x;
    if (t >= N_STEMS * NOUT_STEM) return;
    int s = t / NOUT_STEM, c = t % NOUT_STEM;
    const float* xr = hid + (size_t)s * DIM;
    const float* wr = w + (size_t)c * DIM;
    float acc = b[c];
#pragma unroll 8
    for (int i = 0; i < DIM; ++i) acc += xr[i] * wr[i];
    out[t] = acc;
}

__global__ void jb_out_kernel(const float* __restrict__ hid, const float* __restrict__ w,
                              const float* __restrict__ b, float* __restrict__ out) {
    int t = blockIdx.x * blockDim.x + threadIdx.x;
    if (t >= N_JBONDS) return;
    const float* x0 = hid + (size_t)(2 * t) * DIM;
    const float* x1 = hid + (size_t)(2 * t + 1) * DIM;
    float a0 = b[0], a1 = b[0];
#pragma unroll 8
    for (int i = 0; i < DIM; ++i) { a0 += x0[i] * w[i]; a1 += x1[i] * w[i]; }
    out[t] = 0.5f * (a0 + a1);
}

// ================= Set2Set (1 step, zero-initialized LSTM => q shared by all graphs) ====
__global__ void q0_kernel(const float* __restrict__ bih, const float* __restrict__ bhh,
                          float* __restrict__ q0) {
    int o = threadIdx.x;
    if (o >= DIM) return;
    float gi = bih[o]        + bhh[o];         // i gate
    float gg = bih[128 + o]  + bhh[128 + o];   // g gate
    float go = bih[192 + o]  + bhh[192 + o];   // o gate
    float c = sigm(gi) * tanhf(gg);            // f*c_prev = 0
    q0[o] = sigm(go) * tanhf(c);
}

__global__ void e_kernel(const float* __restrict__ nodeF, const float* __restrict__ q0,
                         const int* __restrict__ batch, float* __restrict__ eN,
                         unsigned* __restrict__ emaxU) {
    int nn = blockIdx.x * blockDim.x + threadIdx.x;
    if (nn >= N_NODES) return;
    const float* xr = nodeF + (size_t)nn * DIM;
    float acc = 0.f;
#pragma unroll 8
    for (int i = 0; i < DIM; ++i) acc += xr[i] * q0[i];
    eN[nn] = acc;
    atomicMax(&emaxU[batch[nn]], f2ord(acc));
}

__global__ void a_kernel(const int* __restrict__ batch, const unsigned* __restrict__ emaxU,
                         float* __restrict__ eN, float* __restrict__ asum) {
    int nn = blockIdx.x * blockDim.x + threadIdx.x;
    if (nn >= N_NODES) return;
    int g = batch[nn];
    float a = expf(eN[nn] - ord2f(emaxU[g]));
    eN[nn] = a;
    atomicAdd(&asum[g], a);
}

__global__ void rpool_kernel(const float* __restrict__ nodeF, const int* __restrict__ batch,
                             const float* __restrict__ eN, const float* __restrict__ asum,
                             float* __restrict__ rpool) {
    int t = blockIdx.x * blockDim.x + threadIdx.x;
    if (t >= N_NODES * DIM) return;
    int nn = t / DIM, o = t % DIM;
    int g = batch[nn];
    float wgt = eN[nn] / asum[g];
    atomicAdd(&rpool[(size_t)g * DIM + o], wgt * nodeF[t]);
}

__global__ void final_kernel(const float* __restrict__ q0, const float* __restrict__ rpool,
                             const float* __restrict__ loutw, const float* __restrict__ loutb,
                             float* __restrict__ out) {
    int t = blockIdx.x * blockDim.x + threadIdx.x;
    if (t >= NUM_GRAPHS * 2) return;
    int g = t / 2, j = t % 2;
    float acc = loutb[j];
    const float* w = loutw + (size_t)j * 2 * DIM;
    const float* rp = rpool + (size_t)g * DIM;
#pragma unroll 8
    for (int k = 0; k < DIM; ++k) acc += w[k] * q0[k];
#pragma unroll 8
    for (int k = 0; k < DIM; ++k) acc += w[DIM + k] * rp[k];
    out[t] = acc;
}

// ================= launch =================
extern "C" void kernel_launch(void* const* d_in, const int* in_sizes, int n_in,
                              void* d_out, int out_size, void* d_ws, size_t ws_size,
                              hipStream_t stream) {
    (void)in_sizes; (void)n_in; (void)out_size; (void)ws_size;
    const float* x         = (const float*)d_in[0];
    const float* edge_attr = (const float*)d_in[1];
    const float* lin0_w    = (const float*)d_in[2];
    const float* lin0_b    = (const float*)d_in[3];
    const float* en1_w     = (const float*)d_in[4];
    const float* en1_b     = (const float*)d_in[5];
    const float* en2_w     = (const float*)d_in[6];
    const float* en2_b     = (const float*)d_in[7];
    const float* root_w    = (const float*)d_in[8];
    const float* conv_b    = (const float*)d_in[9];
    const float* gru_w_ih  = (const float*)d_in[10];
    const float* gru_w_hh  = (const float*)d_in[11];
    const float* gru_b_ih  = (const float*)d_in[12];
    const float* gru_b_hh  = (const float*)d_in[13];
    const float* s1_w = (const float*)d_in[14];
    const float* s1_b = (const float*)d_in[15];
    const float* s2_w = (const float*)d_in[16];
    const float* s2_b = (const float*)d_in[17];
    const float* j1_w = (const float*)d_in[18];
    const float* j1_b = (const float*)d_in[19];
    const float* j2_w = (const float*)d_in[20];
    const float* j2_b = (const float*)d_in[21];
    const float* lstm_b_ih = (const float*)d_in[24];
    const float* lstm_b_hh = (const float*)d_in[25];
    const float* lout_w    = (const float*)d_in[26];
    const float* lout_b    = (const float*)d_in[27];
    const int* edge_index  = (const int*)d_in[28];
    const int* stem_idx    = (const int*)d_in[29];
    const int* jb_idx      = (const int*)d_in[30];
    const int* batch       = (const int*)d_in[31];
    float* out = (float*)d_out;

    // ---- workspace layout ----
    char* base = (char*)d_ws;
    size_t off = 0;
    auto alloc = [&](size_t bytes) { size_t r = off; off = (off + bytes + 255) & ~(size_t)255; return r; };
    float*  nodeF   = (float*) (base + alloc((size_t)N_NODES * DIM * 4));
    bf16_t* nodeBf  = (bf16_t*)(base + alloc((size_t)N_NODES * DIM * 2));
    bf16_t* mBf     = (bf16_t*)(base + alloc((size_t)N_NODES * DIM * 2));
    bf16_t* heBf    = (bf16_t*)(base + alloc((size_t)N_EDGES * DIM * 2));
    bf16_t* en2wBf  = (bf16_t*)(base + alloc((size_t)DIM * DIM * DIM * 2));
    bf16_t* wihBf   = (bf16_t*)(base + alloc((size_t)3 * DIM * DIM * 2));
    bf16_t* whhBf   = (bf16_t*)(base + alloc((size_t)3 * DIM * DIM * 2));
    float*  agg     = (float*) (base + alloc((size_t)N_NODES * DIM * 4));
    float*  deg     = (float*) (base + alloc((size_t)N_NODES * 4));
    float*  stemHid = (float*) (base + alloc((size_t)N_STEMS * DIM * 4));
    float*  jbHid   = (float*) (base + alloc((size_t)2 * N_JBONDS * DIM * 4));
    float*  q0      = (float*) (base + alloc(DIM * 4));
    float*  eN      = (float*) (base + alloc((size_t)N_NODES * 4));
    unsigned* emaxU = (unsigned*)(base + alloc((size_t)NUM_GRAPHS * 4));
    float*  asum    = (float*) (base + alloc((size_t)NUM_GRAPHS * 4));
    float*  rpool   = (float*) (base + alloc((size_t)NUM_GRAPHS * DIM * 4));

    const int TPB = 256;
    auto blocks = [](long n, int tpb) { return (int)((n + tpb - 1) / tpb); };

    // ---- one-time setup ----
    to_bf16<<<blocks(DIM * DIM * DIM, TPB), TPB, 0, stream>>>(en2_w, en2wBf, DIM * DIM * DIM);
    to_bf16<<<blocks(3 * DIM * DIM, TPB), TPB, 0, stream>>>(gru_w_ih, wihBf, 3 * DIM * DIM);
    to_bf16<<<blocks(3 * DIM * DIM, TPB), TPB, 0, stream>>>(gru_w_hh, whhBf, 3 * DIM * DIM);
    lin0_kernel<<<blocks((long)N_NODES * DIM, TPB), TPB, 0, stream>>>(x, lin0_w, lin0_b, nodeF, nodeBf);
    he_kernel<<<blocks((long)N_EDGES * DIM, TPB), TPB, 0, stream>>>(edge_attr, en1_w, en1_b, heBf);
    zero_f32<<<blocks(N_NODES, TPB), TPB, 0, stream>>>(deg, N_NODES);
    deg_kernel<<<blocks(N_EDGES, TPB), TPB, 0, stream>>>(edge_index, deg);

    // ---- 6 message-passing steps ----
    const int etiles = N_EDGES / 16;            // 1875, exact
    const int ntiles = (N_NODES + 15) / 16;     // 938
    for (int step = 0; step < 6; ++step) {
        zero_f32<<<blocks((long)N_NODES * DIM, TPB), TPB, 0, stream>>>(agg, N_NODES * DIM);
        msg_kernel<<<(etiles + 3) / 4, 128, 0, stream>>>(heBf, en2wBf, en2_b, nodeF, edge_index, agg);
        m_kernel<<<blocks((long)N_NODES * DIM, TPB), TPB, 0, stream>>>(agg, deg, nodeF, root_w, conv_b, mBf);
        gru_kernel<<<(ntiles + 3) / 4, 128, 0, stream>>>(mBf, nodeBf, wihBf, whhBf,
                                                         gru_b_ih, gru_b_hh, nodeF, nodeBf);
    }

    // ---- heads ----
    head_hid_kernel<<<blocks((long)N_STEMS * DIM, TPB), TPB, 0, stream>>>(nodeF, stem_idx, s1_w, s1_b, stemHid, N_STEMS);
    stem_out_kernel<<<blocks((long)N_STEMS * NOUT_STEM, TPB), TPB, 0, stream>>>(stemHid, s2_w, s2_b, out + NUM_GRAPHS * 2);
    head_hid_kernel<<<blocks((long)2 * N_JBONDS * DIM, TPB), TPB, 0, stream>>>(nodeF, jb_idx, j1_w, j1_b, jbHid, 2 * N_JBONDS);
    jb_out_kernel<<<blocks(N_JBONDS, TPB), TPB, 0, stream>>>(jbHid, j2_w, j2_b,
                                                             out + NUM_GRAPHS * 2 + (long)N_STEMS * NOUT_STEM);

    // ---- Set2Set + final ----
    q0_kernel<<<1, 64, 0, stream>>>(lstm_b_ih, lstm_b_hh, q0);
    zero_f32<<<blocks(NUM_GRAPHS, TPB), TPB, 0, stream>>>((float*)emaxU, NUM_GRAPHS);
    e_kernel<<<blocks(N_NODES, TPB), TPB, 0, stream>>>(nodeF, q0, batch, eN, emaxU);
    zero_f32<<<blocks(NUM_GRAPHS, TPB), TPB, 0, stream>>>(asum, NUM_GRAPHS);
    a_kernel<<<blocks(N_NODES, TPB), TPB, 0, stream>>>(batch, emaxU, eN, asum);
    zero_f32<<<blocks(NUM_GRAPHS * DIM, TPB), TPB, 0, stream>>>(rpool, NUM_GRAPHS * DIM);
    rpool_kernel<<<blocks((long)N_NODES * DIM, TPB), TPB, 0, stream>>>(nodeF, batch, eN, asum, rpool);
    final_kernel<<<blocks(NUM_GRAPHS * 2, TPB), TPB, 0, stream>>>(q0, rpool, lout_w, lout_b, out);
}
